// Mamba2CriticNetwork_21947282882850
// MI455X (gfx1250) — matrix operand order repaired and checked by reading, not compiled, not run on hardware
//
#include <hip/hip_runtime.h>
#include <hip/hip_bf16.h>

typedef __attribute__((ext_vector_type(16))) _Float16 v16h;
typedef __attribute__((ext_vector_type(8)))  _Float16 h8;
typedef __attribute__((ext_vector_type(8)))  float    v8f;

#if __has_builtin(__builtin_amdgcn_global_load_async_to_lds_b128)
#define HAVE_ASYNC_LDS 1
typedef int gnu_i4 __attribute__((vector_size(4 * sizeof(int))));
typedef __attribute__((address_space(3))) gnu_i4* lds_i4p;
__device__ __forceinline__ void async_g2l_16B(const _Float16* g, _Float16* l){
  __builtin_amdgcn_global_load_async_to_lds_b128(
      (gnu_i4*)(void*)g, (lds_i4p)(void*)l, 0, 0);
}
__device__ __forceinline__ void wait_async0(){
#if __has_builtin(__builtin_amdgcn_s_wait_asynccnt)
  __builtin_amdgcn_s_wait_asynccnt(0);
#else
  asm volatile("s_wait_asynccnt 0" ::: "memory");
#endif
}
#else
#define HAVE_ASYNC_LDS 0
#endif

// ---------------------------------------------------------------------------
// f32 -> f16 convert (weights, once per call)
// ---------------------------------------------------------------------------
__global__ void k_cvt(const float* __restrict__ in, _Float16* __restrict__ out, int n){
  int i = blockIdx.x * 256 + threadIdx.x;
  if (i < n) out[i] = (_Float16)in[i];
}

// ---------------------------------------------------------------------------
// Blocked WMMA GEMM: D[M,N] = act( A[M,K](f16) * W[N,K]^T(f16) + bias )
// Block = 128 threads (4 waves) computing a 64x64 tile.
// Per K-step (32): the 64x32 W tile (4KB f16) is staged into LDS once
// (async-to-LDS when available), then each wave reuses one A fragment
// across 4 WMMAs whose B fragments come from LDS.
// Requires M%64==0, N%64==0, K%32==0.  Fragment layouts per ISA 7.12.2.
// ---------------------------------------------------------------------------
__global__ void k_gemm_wmma_tile(const _Float16* __restrict__ A, const _Float16* __restrict__ W,
                                 const float* __restrict__ bias, void* __restrict__ D,
                                 int M, int N, int K, int lda, int ldw, int ldd,
                                 int relu, int out16)
{
  __shared__ _Float16 wtile[64 * 32];            // [n_local][k_local], 4 KB

  int tiles_n = N >> 6;
  int bm = blockIdx.x / tiles_n;
  int bn = blockIdx.x - bm * tiles_n;
  (void)M;

  int tid  = threadIdx.x;                        // 0..127
  int wv   = tid >> 5;                           // wave in block: M sub-tile
  int lane = tid & 31;
  int g    = lane >> 4;                          // lane-half
  int mr   = lane & 15;

  // cooperative W-tile staging assignment: 2 x 16B chunks per thread
  int srow  = tid >> 1;                          // 0..63
  int spart = tid & 1;                           // 0/1 -> k-halves 0..15 / 16..31
  const _Float16* wsrc = W + (size_t)(bn * 64 + srow) * ldw + spart * 16;
  _Float16* ldst = &wtile[srow * 32 + spart * 16];

  const _Float16* Arow = A + (size_t)(bm * 64 + wv * 16 + mr) * lda;

  v8f acc0 = {0.f,0.f,0.f,0.f,0.f,0.f,0.f,0.f};
  v8f acc1 = acc0, acc2 = acc0, acc3 = acc0;

  for (int kk = 0; kk < K; kk += 32) {
    __syncthreads();                             // previous tile fully consumed
#if HAVE_ASYNC_LDS
    async_g2l_16B(wsrc + kk,     ldst);
    async_g2l_16B(wsrc + kk + 8, ldst + 8);
    wait_async0();
#else
    h8 t0 = *(const h8*)(wsrc + kk);
    h8 t1 = *(const h8*)(wsrc + kk + 8);
    *(h8*)ldst       = t0;
    *(h8*)(ldst + 8) = t1;
#endif
    __syncthreads();                             // tile visible to all waves

    // A 16x32 f16 fragment: half g holds K g*8..g*8+7 and 16+g*8..16+g*8+7
    h8 a0 = *(const h8*)(Arow + kk + (g << 3));
    h8 a1 = *(const h8*)(Arow + kk + 16 + (g << 3));
    v16h av;
#pragma unroll
    for (int i = 0; i < 8; ++i) { av[i] = a0[i]; av[8+i] = a1[i]; }

#pragma unroll
    for (int sub = 0; sub < 4; ++sub) {
      const _Float16* bp = &wtile[(sub * 16 + mr) * 32 + (g << 3)];
      h8 b0 = *(const h8*)bp;
      h8 b1 = *(const h8*)(bp + 16);
      v16h bv;
#pragma unroll
      for (int i = 0; i < 8; ++i) { bv[i] = b0[i]; bv[8+i] = b1[i]; }
      if      (sub == 0) acc0 = __builtin_amdgcn_wmma_f32_16x16x32_f16(false, av, false, bv, (short)0, acc0, false, false);
      else if (sub == 1) acc1 = __builtin_amdgcn_wmma_f32_16x16x32_f16(false, av, false, bv, (short)0, acc1, false, false);
      else if (sub == 2) acc2 = __builtin_amdgcn_wmma_f32_16x16x32_f16(false, av, false, bv, (short)0, acc2, false, false);
      else               acc3 = __builtin_amdgcn_wmma_f32_16x16x32_f16(false, av, false, bv, (short)0, acc3, false, false);
    }
  }

#pragma unroll
  for (int sub = 0; sub < 4; ++sub) {
    v8f acc = (sub == 0) ? acc0 : (sub == 1) ? acc1 : (sub == 2) ? acc2 : acc3;
    int n = bn * 64 + sub * 16 + mr;
    float bv2 = bias ? bias[n] : 0.f;
#pragma unroll
    for (int r = 0; r < 8; ++r) {
      int m = bm * 64 + wv * 16 + (g << 3) + r;  // D: VGPR r -> M = r + 8*half
      float v = acc[r] + bv2;
      if (relu) v = fmaxf(v, 0.f);
      if (out16) ((_Float16*)D)[(size_t)m * ldd + n] = (_Float16)v;
      else       ((float*)   D)[(size_t)m * ldd + n] = v;
    }
  }
}

// ---------------------------------------------------------------------------
// Simple per-wave WMMA GEMM (fallback for small N, e.g. N=32)
// ---------------------------------------------------------------------------
__global__ void k_gemm_wmma(const _Float16* __restrict__ A, const _Float16* __restrict__ W,
                            const float* __restrict__ bias, void* __restrict__ D,
                            int M, int N, int K, int lda, int ldw, int ldd,
                            int relu, int out16)
{
  int wave = blockIdx.x * (blockDim.x >> 5) + (threadIdx.x >> 5);
  int tiles_n = N >> 4;
  int total   = (M >> 4) * tiles_n;
  if (wave >= total) return;            // uniform per-wave: EXEC stays all-1s for WMMA
  int mt = wave / tiles_n;
  int nt = wave - mt * tiles_n;
  int lane = threadIdx.x & 31;
  int g  = lane >> 4;
  int mr = lane & 15;

  const _Float16* Arow = A + (size_t)(mt * 16 + mr) * lda;
  const _Float16* Wrow = W + (size_t)(nt * 16 + mr) * ldw;

  v8f acc = {0.f,0.f,0.f,0.f,0.f,0.f,0.f,0.f};
  for (int kk = 0; kk < K; kk += 32) {
    h8 a0 = *(const h8*)(Arow + kk + (g << 3));
    h8 a1 = *(const h8*)(Arow + kk + 16 + (g << 3));
    h8 b0 = *(const h8*)(Wrow + kk + (g << 3));
    h8 b1 = *(const h8*)(Wrow + kk + 16 + (g << 3));
    v16h av, bv;
#pragma unroll
    for (int i = 0; i < 8; ++i) { av[i] = a0[i]; av[8+i] = a1[i]; bv[i] = b0[i]; bv[8+i] = b1[i]; }
    acc = __builtin_amdgcn_wmma_f32_16x16x32_f16(false, av, false, bv,
                                                 (short)0, acc, false, false);
  }
  int n = (nt << 4) + mr;
  float bv2 = bias ? bias[n] : 0.f;
#pragma unroll
  for (int r = 0; r < 8; ++r) {
    int m = (mt << 4) + (g << 3) + r;
    float v = acc[r] + bv2;
    if (relu) v = fmaxf(v, 0.f);
    if (out16) ((_Float16*)D)[(size_t)m * ldd + n] = (_Float16)v;
    else       ((float*)   D)[(size_t)m * ldd + n] = v;
  }
}

// ---------------------------------------------------------------------------
// Scan encoder conv1: (B,1,1080) -> (B,32,540), k=5 s=2 pad=2, relu; input clip/10
// ---------------------------------------------------------------------------
__global__ void k_conv1(const float* __restrict__ scan, const float* __restrict__ w,
                        const float* __restrict__ b, float* __restrict__ c1){
  int idx = blockIdx.x * 256 + threadIdx.x;
  if (idx >= 1024 * 32 * 540) return;
  int p  = idx % 540;
  int t  = idx / 540;
  int oc = t % 32;
  int bb = t / 32;
  const float* s = scan + (size_t)bb * 1080;
  float acc = b[oc];
#pragma unroll
  for (int k = 0; k < 5; ++k) {
    int pos = 2 * p + k - 2;
    if (pos >= 0 && pos < 1080) {
      float v = fminf(fmaxf(s[pos], 0.f), 10.f) * 0.1f;
      acc += w[oc * 5 + k] * v;
    }
  }
  c1[idx] = fmaxf(acc, 0.f);
}

// conv2: (B,32,540) -> (B,64,270), k=5 s=2 pad=2, relu; flattened f16 (B,17280)
__global__ void k_conv2(const float* __restrict__ c1, const float* __restrict__ w,
                        const float* __restrict__ b, _Float16* __restrict__ outp){
  int idx = blockIdx.x * 256 + threadIdx.x;
  if (idx >= 1024 * 64 * 270) return;
  int q  = idx % 270;
  int t  = idx / 270;
  int oc = t % 64;
  int bb = t / 64;
  const float* cin = c1 + (size_t)bb * 32 * 540;
  float acc = b[oc];
  for (int ic = 0; ic < 32; ++ic) {
    const float* row = cin + ic * 540;
    const float* wr  = w + (oc * 32 + ic) * 5;
#pragma unroll
    for (int k = 0; k < 5; ++k) {
      int pos = 2 * q + k - 2;
      if (pos >= 0 && pos < 540) acc += wr[k] * row[pos];
    }
  }
  outp[(size_t)bb * 17280 + oc * 270 + q] = (_Float16)fmaxf(acc, 0.f);
}

// ---------------------------------------------------------------------------
// Odom branch: normalize state, linear 4->64, write into comb[:,256:320] (f16)
// ---------------------------------------------------------------------------
__global__ void k_odom(const float* __restrict__ state, const float* __restrict__ w,
                       const float* __restrict__ b, _Float16* __restrict__ comb){
  int idx = blockIdx.x * 256 + threadIdx.x;
  if (idx >= 1024 * 64) return;
  int j  = idx & 63;
  int bb = idx >> 6;
  const float lo[4] = {-5.f, -5.f, -15.f, -10.f};
  const float hi[4] = {20.f,  5.f,  15.f,  10.f};
  float acc = b[j];
#pragma unroll
  for (int i = 0; i < 4; ++i) {
    float sn = 2.f * (state[bb * 4 + i] - lo[i]) / (hi[i] - lo[i] + 1e-8f) - 1.f;
    sn = fminf(fmaxf(sn, -1.f), 1.f);
    acc += w[j * 4 + i] * sn;
  }
  comb[(size_t)bb * 320 + 256 + j] = (_Float16)acc;
}

// ---------------------------------------------------------------------------
// buf assembly + pre-LN.  One block per (t,b) row (t-major), 256 threads.
// buf (output, (B,48,256) f32) ; u = LN(buf) stored f16 t-major (48,1024,256)
// ---------------------------------------------------------------------------
__global__ void k_buf_ln(const float* __restrict__ obs, const float* __restrict__ proj,
                         const float* __restrict__ g, const float* __restrict__ be,
                         float* __restrict__ buf_out, _Float16* __restrict__ u){
  int r = blockIdx.x;          // 0..49151
  int t = r >> 10;
  int b = r & 1023;
  int d = threadIdx.x;
  float v = (t < 47) ? obs[((size_t)b * 48 + (t + 1)) * 256 + d]
                     : proj[(size_t)b * 256 + d];
  buf_out[((size_t)b * 48 + t) * 256 + d] = v;

  float s = v, s2 = v * v;
#pragma unroll
  for (int m = 16; m >= 1; m >>= 1) { s += __shfl_xor(s, m, 32); s2 += __shfl_xor(s2, m, 32); }
  __shared__ float sh1[8], sh2[8];
  int w = threadIdx.x >> 5, l = threadIdx.x & 31;
  if (l == 0) { sh1[w] = s; sh2[w] = s2; }
  __syncthreads();
  if (w == 0) {
    float a  = (l < 8) ? sh1[l] : 0.f;
    float a2 = (l < 8) ? sh2[l] : 0.f;
#pragma unroll
    for (int m = 4; m >= 1; m >>= 1) { a += __shfl_xor(a, m, 32); a2 += __shfl_xor(a2, m, 32); }
    if (l == 0) { sh1[0] = a; sh2[0] = a2; }
  }
  __syncthreads();
  float mean = sh1[0] * (1.f / 256.f);
  float var  = sh2[0] * (1.f / 256.f) - mean * mean;
  float nv = (v - mean) * rsqrtf(var + 1e-5f) * g[d] + be[d];
  u[(size_t)r * 256 + d] = (_Float16)nv;
}

// ---------------------------------------------------------------------------
// Depthwise causal conv (k=4, pad 3 left) + SiLU over xBC (544 ch) and
// softplus(dt + bias) for the 32 dt channels.  In/out layout (48,1024,576).
// ---------------------------------------------------------------------------
__global__ void k_dtconv(const float* __restrict__ xin, const float* __restrict__ cw,
                         const float* __restrict__ cb, const float* __restrict__ dtb,
                         float* __restrict__ xout){
  int idx = blockIdx.x * 256 + threadIdx.x;
  if (idx >= 48 * 1024 * 576) return;
  int c   = idx % 576;
  int row = idx / 576;
  int t   = row >> 10;
  int b   = row & 1023;
  float o;
  if (c < 544) {
    float acc = cb[c];
#pragma unroll
    for (int k = 0; k < 4; ++k) {
      int ts = t - 3 + k;
      if (ts >= 0) acc += cw[c * 4 + k] * xin[((size_t)(ts << 10 | b)) * 576 + c];
    }
    o = acc / (1.f + expf(-acc));                       // silu
  } else {
    float v = xin[(size_t)row * 576 + c] + dtb[c - 544];
    o = (v > 20.f) ? v : log1pf(expf(v));               // softplus
  }
  xout[(size_t)row * 576 + c] = o;
}

// ---------------------------------------------------------------------------
// Mamba2 SSM scan.  One wave32 per (batch, head): state 16x16 = 8 VGPR/lane.
// lane: n = lane&15 (d_state), half g = lane>>4 ; reg r -> p = r + 8g (head dim)
// Only the last timestep's y is needed downstream.
// ---------------------------------------------------------------------------
__global__ void k_scan(const float* __restrict__ xcdt, const float* __restrict__ A_log,
                       const float* __restrict__ Dv, float* __restrict__ y){
  int wid  = (blockIdx.x << 3) + (threadIdx.x >> 5);
  if (wid >= 1024 * 32) return;
  int lane = threadIdx.x & 31;
  int b = wid >> 5;
  int h = wid & 31;
  float Ah = -expf(A_log[h]);
  float Dh = Dv[h];
  int g = lane >> 4, n = lane & 15;
  float hst[8] = {0.f,0.f,0.f,0.f,0.f,0.f,0.f,0.f};
  float xl[8]  = {0.f,0.f,0.f,0.f,0.f,0.f,0.f,0.f};
  float Cn = 0.f;
  for (int t = 0; t < 48; ++t) {
    const float* base = xcdt + ((size_t)((t << 10) | b)) * 576;
    float dtv = base[544 + h];
    float dA  = expf(dtv * Ah);
    float Bn  = base[512 + n];
#pragma unroll
    for (int r = 0; r < 8; ++r) {
      int p = r + (g << 3);
      float xv = base[h * 16 + p];
      hst[r] = hst[r] * dA + (dtv * xv) * Bn;
      if (t == 47) xl[r] = xv;
    }
    if (t == 47) Cn = base[528 + n];
  }
#pragma unroll
  for (int r = 0; r < 8; ++r) {
    float s = hst[r] * Cn;                 // reduce over n (16 lanes per half)
    s += __shfl_xor(s, 1, 32);
    s += __shfl_xor(s, 2, 32);
    s += __shfl_xor(s, 4, 32);
    s += __shfl_xor(s, 8, 32);
    if (n == 0) {
      int p = r + (g << 3);
      y[(size_t)b * 512 + h * 16 + p] = s + Dh * xl[r];
    }
  }
}

// ---------------------------------------------------------------------------
// y * silu(z) then RMS-norm over 512, f16 out.  One block per batch row.
// ---------------------------------------------------------------------------
__global__ void k_gate_rms(const float* __restrict__ y, const float* __restrict__ z,
                           const float* __restrict__ rg, _Float16* __restrict__ yn){
  int b = blockIdx.x;
  int d = threadIdx.x;
  float vs[2];
  float acc = 0.f;
#pragma unroll
  for (int i = 0; i < 2; ++i) {
    int dd = d + i * 256;
    float zv = z[(size_t)b * 512 + dd];
    float yv = y[(size_t)b * 512 + dd] * (zv / (1.f + expf(-zv)));
    vs[i] = yv; acc += yv * yv;
  }
#pragma unroll
  for (int m = 16; m >= 1; m >>= 1) acc += __shfl_xor(acc, m, 32);
  __shared__ float sh[8];
  int w = threadIdx.x >> 5, l = threadIdx.x & 31;
  if (l == 0) sh[w] = acc;
  __syncthreads();
  if (w == 0) {
    float a = (l < 8) ? sh[l] : 0.f;
#pragma unroll
    for (int m = 4; m >= 1; m >>= 1) a += __shfl_xor(a, m, 32);
    if (l == 0) sh[0] = a;
  }
  __syncthreads();
  float inv = rsqrtf(sh[0] * (1.f / 512.f) + 1e-5f);
#pragma unroll
  for (int i = 0; i < 2; ++i) {
    int dd = d + i * 256;
    yn[(size_t)b * 512 + dd] = (_Float16)(vs[i] * inv * rg[dd]);
  }
}

// ---------------------------------------------------------------------------
// LayerNorm over 256 (feat = LN(mo_last)), f16 out. One block per batch row.
// ---------------------------------------------------------------------------
__global__ void k_ln_feat(const float* __restrict__ mo, const float* __restrict__ g,
                          const float* __restrict__ be, _Float16* __restrict__ feat){
  int b = blockIdx.x;
  int d = threadIdx.x;
  float v = mo[(size_t)b * 256 + d];
  float s = v, s2 = v * v;
#pragma unroll
  for (int m = 16; m >= 1; m >>= 1) { s += __shfl_xor(s, m, 32); s2 += __shfl_xor(s2, m, 32); }
  __shared__ float sh1[8], sh2[8];
  int w = threadIdx.x >> 5, l = threadIdx.x & 31;
  if (l == 0) { sh1[w] = s; sh2[w] = s2; }
  __syncthreads();
  if (w == 0) {
    float a  = (l < 8) ? sh1[l] : 0.f;
    float a2 = (l < 8) ? sh2[l] : 0.f;
#pragma unroll
    for (int m = 4; m >= 1; m >>= 1) { a += __shfl_xor(a, m, 32); a2 += __shfl_xor(a2, m, 32); }
    if (l == 0) { sh1[0] = a; sh2[0] = a2; }
  }
  __syncthreads();
  float mean = sh1[0] * (1.f / 256.f);
  float var  = sh2[0] * (1.f / 256.f) - mean * mean;
  feat[(size_t)b * 256 + d] = (_Float16)((v - mean) * rsqrtf(var + 1e-5f) * g[d] + be[d]);
}

// value head: (1024,32) @ (32,)^T + b
__global__ void k_fc3(const float* __restrict__ v2, const float* __restrict__ w,
                      const float* __restrict__ bb, float* __restrict__ out){
  int b = blockIdx.x * 256 + threadIdx.x;
  if (b >= 1024) return;
  float acc = bb[0];
#pragma unroll
  for (int j = 0; j < 32; ++j) acc += v2[(size_t)b * 32 + j] * w[j];
  out[b] = acc;
}

// ---------------------------------------------------------------------------
static void gemm(hipStream_t s, const _Float16* A, const _Float16* W, const float* bias,
                 void* D, int M, int N, int K, int lda, int ldw, int ldd,
                 int relu, int out16)
{
  if ((M % 64 == 0) && (N % 64 == 0)) {
    int blocks = (M / 64) * (N / 64);
    k_gemm_wmma_tile<<<blocks, 128, 0, s>>>(A, W, bias, D, M, N, K, lda, ldw, ldd, relu, out16);
  } else {
    int tiles = (M / 16) * (N / 16);
    int blocks = (tiles + 3) / 4;
    k_gemm_wmma<<<blocks, 128, 0, s>>>(A, W, bias, D, M, N, K, lda, ldw, ldd, relu, out16);
  }
}

extern "C" void kernel_launch(void* const* d_in, const int* in_sizes, int n_in,
                              void* d_out, int out_size, void* d_ws, size_t ws_size,
                              hipStream_t stream)
{
  (void)in_sizes; (void)n_in; (void)out_size; (void)ws_size;

  const float* scan     = (const float*)d_in[0];
  const float* state    = (const float*)d_in[1];
  const float* obs      = (const float*)d_in[2];
  const float* conv1_w  = (const float*)d_in[3];
  const float* conv1_b  = (const float*)d_in[4];
  const float* conv2_w  = (const float*)d_in[5];
  const float* conv2_b  = (const float*)d_in[6];
  const float* enc_fc_w = (const float*)d_in[7];
  const float* enc_fc_b = (const float*)d_in[8];
  const float* odom_w   = (const float*)d_in[9];
  const float* odom_b   = (const float*)d_in[10];
  const float* proj1_w  = (const float*)d_in[11];
  const float* proj1_b  = (const float*)d_in[12];
  const float* proj2_w  = (const float*)d_in[13];
  const float* proj2_b  = (const float*)d_in[14];
  const float* pre_g    = (const float*)d_in[15];
  const float* pre_b    = (const float*)d_in[16];
  const float* in_w     = (const float*)d_in[17];
  const float* cw       = (const float*)d_in[18];
  const float* cb       = (const float*)d_in[19];
  const float* dtb      = (const float*)d_in[20];
  const float* A_log    = (const float*)d_in[21];
  const float* Dvec     = (const float*)d_in[22];
  const float* rms_g    = (const float*)d_in[23];
  const float* out_w    = (const float*)d_in[24];
  const float* ln_g     = (const float*)d_in[25];
  const float* ln_b     = (const float*)d_in[26];
  const float* fc1_w    = (const float*)d_in[27];
  const float* fc1_b    = (const float*)d_in[28];
  const float* fc2_w    = (const float*)d_in[29];
  const float* fc2_b    = (const float*)d_in[30];
  const float* fc3_w    = (const float*)d_in[31];
  const float* fc3_b    = (const float*)d_in[32];

  float* value   = (float*)d_out;          // (1024,1)
  float* buf_out = (float*)d_out + 1024;   // (1024,48,256)

  char* base = (char*)d_ws;
  size_t cur = 0;
  auto alloc = [&](size_t bytes) -> void* {
    cur = (cur + 255) & ~(size_t)255;
    void* p = base + cur;
    cur += bytes;
    return p;
  };

  // -- region A (recycled for xBCdt after the scan encoder is consumed) -----
  float*    c1       = (float*)   alloc((size_t)1024 * 32 * 540 * 4);   // 70.8 MB
  _Float16* conv_out = (_Float16*)alloc((size_t)1024 * 17280 * 2);      // 35.4 MB
  _Float16* enc_w16  = (_Float16*)alloc((size_t)17280 * 256 * 2);       //  8.8 MB
  float*    xBCdt    = (float*)d_ws;  // reuses region A: needs 113.2 MB <= 115.0 MB

  // -- persistent f16 weights ----------------------------------------------
  _Float16* in_w16  = (_Float16*)alloc((size_t)1088 * 256 * 2);
  _Float16* p1_w16  = (_Float16*)alloc((size_t)768 * 320 * 2);
  _Float16* p2_w16  = (_Float16*)alloc((size_t)256 * 768 * 2);
  _Float16* out_w16 = (_Float16*)alloc((size_t)256 * 512 * 2);
  _Float16* fc1_w16 = (_Float16*)alloc((size_t)256 * 256 * 2);
  _Float16* fc2_w16 = (_Float16*)alloc((size_t)32 * 256 * 2);

  // -- activations ----------------------------------------------------------
  _Float16* comb = (_Float16*)alloc((size_t)1024 * 320 * 2);
  _Float16* h1   = (_Float16*)alloc((size_t)1024 * 768 * 2);
  float*    proj = (float*)   alloc((size_t)1024 * 256 * 4);
  _Float16* u    = (_Float16*)alloc((size_t)49152 * 256 * 2);           // (48,1024,256) t-major
  float*    xcdt = (float*)   alloc((size_t)48 * 1024 * 576 * 4);       // 113.2 MB
  float*    zbuf = (float*)   alloc((size_t)1024 * 512 * 4);
  float*    ybuf = (float*)   alloc((size_t)1024 * 512 * 4);
  _Float16* yn   = (_Float16*)alloc((size_t)1024 * 512 * 2);
  float*    mo   = (float*)   alloc((size_t)1024 * 256 * 4);
  _Float16* feat = (_Float16*)alloc((size_t)1024 * 256 * 2);
  _Float16* v1   = (_Float16*)alloc((size_t)1024 * 256 * 2);
  float*    v2   = (float*)   alloc((size_t)1024 * 32 * 4);

  // -- weight conversions ---------------------------------------------------
  auto cvt = [&](const float* src, _Float16* dst, int n) {
    k_cvt<<<(n + 255) / 256, 256, 0, stream>>>(src, dst, n);
  };
  cvt(enc_fc_w, enc_w16, 17280 * 256);
  cvt(in_w,     in_w16,  1088 * 256);
  cvt(proj1_w,  p1_w16,  768 * 320);
  cvt(proj2_w,  p2_w16,  256 * 768);
  cvt(out_w,    out_w16, 256 * 512);
  cvt(fc1_w,    fc1_w16, 256 * 256);
  cvt(fc2_w,    fc2_w16, 32 * 256);

  // -- scan encoder ---------------------------------------------------------
  k_conv1<<<(1024 * 32 * 540 + 255) / 256, 256, 0, stream>>>(scan, conv1_w, conv1_b, c1);
  k_conv2<<<(1024 * 64 * 270 + 255) / 256, 256, 0, stream>>>(c1, conv2_w, conv2_b, conv_out);
  // enc_fc -> vis, written f16 into comb[:, :256] (ldd = 320), relu
  gemm(stream, conv_out, enc_w16, enc_fc_b, comb, 1024, 256, 17280, 17280, 17280, 320, 1, 1);
  k_odom<<<(1024 * 64 + 255) / 256, 256, 0, stream>>>(state, odom_w, odom_b, comb);

  // -- projection MLP -------------------------------------------------------
  gemm(stream, comb, p1_w16, proj1_b, h1,   1024, 768, 320, 320, 320, 768, 1, 1);
  gemm(stream, h1,   p2_w16, proj2_b, proj, 1024, 256, 768, 768, 768, 256, 1, 0);

  // -- buf assembly + pre-LN (u in t-major (48,1024,256)) -------------------
  k_buf_ln<<<49152, 256, 0, stream>>>(obs, proj, pre_g, pre_b, buf_out, u);

  // -- in_proj: xBC+dt columns for all t ; z columns only for t=47 ----------
  gemm(stream, u, in_w16 + (size_t)512 * 256, nullptr, xBCdt,
       49152, 576, 256, 256, 256, 576, 0, 0);
  gemm(stream, u + (size_t)47 * 1024 * 256, in_w16, nullptr, zbuf,
       1024, 512, 256, 256, 256, 512, 0, 0);

  // -- depthwise conv + silu + softplus(dt) ---------------------------------
  k_dtconv<<<(48 * 1024 * 576 + 255) / 256, 256, 0, stream>>>(xBCdt, cw, cb, dtb, xcdt);

  // -- SSM scan (one wave per (b,h)) ----------------------------------------
  k_scan<<<4096, 256, 0, stream>>>(xcdt, A_log, Dvec, ybuf);

  // -- gate + RMS norm + out_proj (last timestep only) ----------------------
  k_gate_rms<<<1024, 256, 0, stream>>>(ybuf, zbuf, rms_g, yn);
  gemm(stream, yn, out_w16, nullptr, mo, 1024, 256, 512, 512, 512, 256, 0, 0);

  // -- value head -----------------------------------------------------------
  k_ln_feat<<<1024, 256, 0, stream>>>(mo, ln_g, ln_b, feat);
  gemm(stream, feat, fc1_w16, fc1_b, v1, 1024, 256, 256, 256, 256, 256, 1, 1);
  gemm(stream, v1,   fc2_w16, fc2_b, v2, 1024, 32,  256, 256, 256, 32,  1, 0);
  k_fc3<<<4, 256, 0, stream>>>(v2, fc3_w, fc3_b, value);
}
